// SelectiveSSM_86019605004387
// MI455X (gfx1250) — matrix operand order repaired and checked by reading, not compiled, not run on hardware
//
#include <hip/hip_runtime.h>
#include <hip/hip_bf16.h>

#define D_MODELc 1024
#define D_STATEc 16
#define D_INNERc 2048
#define B_SZc    2
#define T_LENc   2048
#define BTc      (B_SZc * T_LENc)      /* 4096 rows */
#define N_XZc    (2 * D_INNERc)        /* 4096 */
#define N_SSMPAD 64                    /* 33 -> 64 padded */

typedef __attribute__((ext_vector_type(16))) __bf16 bf16x16;
typedef __attribute__((ext_vector_type(8)))  __bf16 bf16x8;
typedef __attribute__((ext_vector_type(8)))  float  f32x8;

union ABFrag { bf16x16 v; bf16x8 h[2]; };

__device__ __forceinline__ unsigned short f2bf(float f) {
    union { float f; unsigned int u; } c; c.f = f;
    unsigned int u = c.u;
    unsigned int r = u + 0x7FFFu + ((u >> 16) & 1u);   // round-to-nearest-even
    return (unsigned short)(r >> 16);
}

__device__ __forceinline__ float fsigmoid(float x) { return 1.f / (1.f + __expf(-x)); }

// ---------------- conversion kernels ----------------
__global__ void k_f32_to_bf16(const float* __restrict__ src,
                              unsigned short* __restrict__ dst, int n) {
    int i = blockIdx.x * blockDim.x + threadIdx.x;
    if (i < n) dst[i] = f2bf(src[i]);
}

// W_x (33 x 2048) -> padded bf16 (64 x 2048), rows >=33 zero
__global__ void k_pad_wx(const float* __restrict__ src,
                         unsigned short* __restrict__ dst) {
    int i = blockIdx.x * blockDim.x + threadIdx.x;
    if (i >= N_SSMPAD * D_INNERc) return;
    int nrow = i / D_INNERc;
    dst[i] = (nrow < 2 * D_STATEc + 1) ? f2bf(src[i]) : (unsigned short)0;
}

// ---------------- bf16 WMMA GEMM: C[M,ldc] = A[M,K] * Bt[N,K]^T ----------------
// One wave -> 32-row x 64-col tile (2 M-tiles x 4 N-tiles).
// Requires M%32==0, N%64==0, K%32==0.
__global__ __launch_bounds__(256) void k_gemm_bf16(
    const unsigned short* __restrict__ Au,
    const unsigned short* __restrict__ Bu,
    float* __restrict__ C,
    int M, int N, int K, int ldc)
{
    const __bf16* A  = (const __bf16*)Au;
    const __bf16* Bt = (const __bf16*)Bu;
    const int lane = threadIdx.x & 31;
    const int wave = threadIdx.x >> 5;
    const int gw   = blockIdx.x * 8 + wave;
    const int ntb  = N >> 6;           // N-tile groups of 64
    const int mtl  = M >> 5;           // 32-row blocks
    if (gw >= mtl * ntb) return;       // wave-uniform: EXEC stays all-1s for WMMA
    const int mt  = gw / ntb;
    const int nt0 = (gw % ntb) << 2;   // first of 4 16-col tiles

    const int l16 = lane & 15;
    const int hi  = lane >> 4;

    // A fragments: rows mt*32+l16 and +16; lanes 0-15 hold K {0..7,16..23}, lanes 16-31 {8..15,24..31}
    const __bf16* arow0 = A + (size_t)(mt * 32 + l16) * K + hi * 8;
    const __bf16* arow1 = arow0 + (size_t)16 * K;
    // B fragments: lane holds column (tile*16+l16); lanes 0-15 hold K {0..7,16..23}, lanes 16-31 {8..15,24..31}
    const __bf16* brow = Bt + (size_t)(nt0 * 16 + l16) * K + hi * 8;

    f32x8 acc[2][4] = {};
    for (int kb = 0; kb < K; kb += 32) {
        __builtin_prefetch((const void*)(arow0 + kb + 256), 0, 3);  // global_prefetch_b8
        __builtin_prefetch((const void*)(arow1 + kb + 256), 0, 3);
        // ---- load phase: distinct registers so loads clause together ----
        ABFrag a0, a1, b[4];
        a0.h[0] = *(const bf16x8*)(arow0 + kb);
        a0.h[1] = *(const bf16x8*)(arow0 + kb + 16);
        a1.h[0] = *(const bf16x8*)(arow1 + kb);
        a1.h[1] = *(const bf16x8*)(arow1 + kb + 16);
#pragma unroll
        for (int j = 0; j < 4; ++j) {
            const __bf16* bp = brow + (size_t)j * 16 * K + kb;
            b[j].h[0] = *(const bf16x8*)(bp);
            b[j].h[1] = *(const bf16x8*)(bp + 16);
        }
        // ---- compute phase: 8 WMMAs, each B fragment reused twice ----
#pragma unroll
        for (int j = 0; j < 4; ++j) {
            acc[0][j] = __builtin_amdgcn_wmma_f32_16x16x32_bf16(
                false, a0.v, false, b[j].v, (short)0, acc[0][j], false, false);
            acc[1][j] = __builtin_amdgcn_wmma_f32_16x16x32_bf16(
                false, a1.v, false, b[j].v, (short)0, acc[1][j], false, false);
        }
    }
    // D layout: VGPR r -> row = tile*16 + r + hi*8, col = tile*16 + l16
#pragma unroll
    for (int mi = 0; mi < 2; ++mi) {
#pragma unroll
        for (int j = 0; j < 4; ++j) {
            int col = (nt0 + j) * 16 + l16;
#pragma unroll
            for (int r = 0; r < 8; ++r) {
                int row = mt * 32 + mi * 16 + r + hi * 8;
                C[(size_t)row * ldc + col] = acc[mi][j][r];
            }
        }
    }
}

// ---------------- depthwise causal conv (w=4) + bias + SiLU ----------------
// xz: [BT][4096], x_main = cols 0..2047. Outputs f32 + bf16.
__global__ void k_conv_silu(const float* __restrict__ xz,
                            const float* __restrict__ cw,   // [D_INNER*4]
                            const float* __restrict__ cb,   // [D_INNER]
                            float* __restrict__ xconv,
                            unsigned short* __restrict__ xconvbf)
{
    int i = blockIdx.x * blockDim.x + threadIdx.x;
    if (i >= BTc * D_INNERc) return;
    int d  = i % D_INNERc;
    int bt = i / D_INNERc;
    int t  = bt % T_LENc;
    float s = cb[d];
#pragma unroll
    for (int j = 0; j < 4; ++j) {
        int tt = t - 3 + j;
        if (tt >= 0) s += cw[d * 4 + j] * xz[(size_t)(bt - 3 + j) * N_XZc + d];
    }
    float v = s * fsigmoid(s);
    xconv[i]   = v;
    xconvbf[i] = f2bf(v);
}

// ---------------- selective scan: lane n = state n, 2 channels / wave32 ----------------
__global__ __launch_bounds__(256) void k_scan(
    const float* __restrict__ xssm,   // [BT][64]: B=0..15, C=16..31, dt=32
    const float* __restrict__ xconv,  // [BT][D_INNER]  (u)
    const float* __restrict__ Wdt,    // [D_INNER]
    const float* __restrict__ bdt,    // [D_INNER]
    const float* __restrict__ Dv,     // [D_INNER]
    float* __restrict__ yscan)        // [BT][D_INNER]
{
    const int n  = threadIdx.x & 15;
    const int ch = blockIdx.x * 16 + (threadIdx.x >> 4);   // 0..4095 = (b,d)
    const int b  = ch / D_INNERc;
    const int d  = ch % D_INNERc;
    const float wdt = Wdt[d], bd = bdt[d], Dd = Dv[d];
    const float negA = -(float)(n + 1);
    float h = 0.f;
    const float* srow = xssm  + (size_t)b * T_LENc * N_SSMPAD;
    const float* ucol = xconv + (size_t)b * T_LENc * D_INNERc + d;
    float*       ycol = yscan + (size_t)b * T_LENc * D_INNERc + d;
    for (int t = 0; t < T_LENc; ++t) {
        const float* sr = srow + (size_t)t * N_SSMPAD;
        float dtr = sr[2 * D_STATEc];
        float sp  = (dtr > 20.f) ? dtr : __logf(1.f + __expf(dtr));  // softplus
        float dtp = sp * wdt + bd;
        float An  = __expf(negA * dtp);
        float Bn  = sr[n];
        float Cn  = sr[D_STATEc + n];
        float u   = ucol[(size_t)t * D_INNERc];
        h = An * h + Bn * u;
        float p = Cn * h;
        p += __shfl_xor(p, 1, 16);
        p += __shfl_xor(p, 2, 16);
        p += __shfl_xor(p, 4, 16);
        p += __shfl_xor(p, 8, 16);
        if (n == 0) ycol[(size_t)t * D_INNERc] = p + Dd * u;
    }
}

// ---------------- gate: y * silu(z) -> bf16 ----------------
__global__ void k_gate(const float* __restrict__ xz,
                       const float* __restrict__ yscan,
                       unsigned short* __restrict__ ygbf)
{
    int i = blockIdx.x * blockDim.x + threadIdx.x;
    if (i >= BTc * D_INNERc) return;
    int d  = i % D_INNERc;
    int bt = i / D_INNERc;
    float z = xz[(size_t)bt * N_XZc + D_INNERc + d];
    ygbf[i] = f2bf(yscan[i] * (z * fsigmoid(z)));
}

extern "C" void kernel_launch(void* const* d_in, const int* in_sizes, int n_in,
                              void* d_out, int out_size, void* d_ws, size_t ws_size,
                              hipStream_t stream)
{
    const float* x     = (const float*)d_in[0];
    const float* W_in  = (const float*)d_in[1];
    const float* convw = (const float*)d_in[2];
    const float* convb = (const float*)d_in[3];
    const float* W_x   = (const float*)d_in[4];
    const float* W_dt  = (const float*)d_in[5];
    const float* b_dt  = (const float*)d_in[6];
    const float* Dvec  = (const float*)d_in[7];
    const float* W_out = (const float*)d_in[8];
    float* out = (float*)d_out;

    char* w = (char*)d_ws;
    size_t o = 0;
    auto alloc = [&](size_t bytes) -> char* {
        char* p = w + o; o += (bytes + 255) & ~(size_t)255; return p;
    };
    float*          xz      = (float*)         alloc((size_t)BTc * N_XZc * 4);       // 64MB
    unsigned short* xbf     = (unsigned short*)alloc((size_t)BTc * D_MODELc * 2);    // 8MB
    unsigned short* winbf   = (unsigned short*)alloc((size_t)N_XZc * D_MODELc * 2);  // 8MB
    float*          xconv   = (float*)         alloc((size_t)BTc * D_INNERc * 4);    // 32MB
    unsigned short* xconvbf = (unsigned short*)alloc((size_t)BTc * D_INNERc * 2);    // 16MB
    unsigned short* wxbf    = (unsigned short*)alloc((size_t)N_SSMPAD * D_INNERc * 2);
    float*          xssm    = (float*)         alloc((size_t)BTc * N_SSMPAD * 4);    // 1MB
    float*          yscan   = (float*)         alloc((size_t)BTc * D_INNERc * 4);    // 32MB
    unsigned short* ygbf    = (unsigned short*)alloc((size_t)BTc * D_INNERc * 2);    // 16MB
    unsigned short* woutbf  = (unsigned short*)alloc((size_t)D_MODELc * D_INNERc * 2);

    const int thr = 256;
    k_f32_to_bf16<<<(BTc * D_MODELc + thr - 1) / thr, thr, 0, stream>>>(x, xbf, BTc * D_MODELc);
    k_f32_to_bf16<<<(N_XZc * D_MODELc + thr - 1) / thr, thr, 0, stream>>>(W_in, winbf, N_XZc * D_MODELc);
    k_f32_to_bf16<<<(D_MODELc * D_INNERc + thr - 1) / thr, thr, 0, stream>>>(W_out, woutbf, D_MODELc * D_INNERc);
    k_pad_wx<<<(N_SSMPAD * D_INNERc + thr - 1) / thr, thr, 0, stream>>>(W_x, wxbf);

    { // xz = x @ W_in^T : M=4096 N=4096 K=1024
        int waves = (BTc / 32) * (N_XZc / 64);
        k_gemm_bf16<<<(waves + 7) / 8, 256, 0, stream>>>(xbf, winbf, xz, BTc, N_XZc, D_MODELc, N_XZc);
    }
    k_conv_silu<<<(BTc * D_INNERc + thr - 1) / thr, thr, 0, stream>>>(xz, convw, convb, xconv, xconvbf);
    { // xssm = xconv @ Wx^T (padded N=64) : M=4096 N=64 K=2048
        int waves = (BTc / 32) * (N_SSMPAD / 64);
        k_gemm_bf16<<<(waves + 7) / 8, 256, 0, stream>>>(xconvbf, wxbf, xssm, BTc, N_SSMPAD, D_INNERc, N_SSMPAD);
    }
    k_scan<<<(B_SZc * D_INNERc) / 16, 256, 0, stream>>>(xssm, xconv, W_dt, b_dt, Dvec, yscan);
    k_gate<<<(BTc * D_INNERc + thr - 1) / thr, thr, 0, stream>>>(xz, yscan, ygbf);
    { // out = ygate @ W_out^T : M=4096 N=1024 K=2048
        int waves = (BTc / 32) * (D_MODELc / 64);
        k_gemm_bf16<<<(waves + 7) / 8, 256, 0, stream>>>(ygbf, woutbf, out, BTc, D_MODELc, D_INNERc, D_MODELc);
    }
}
